// l1_l1_43147241455994
// MI455X (gfx1250) — matrix operand order repaired and checked by reading, not compiled, not run on hardware
//
#include <hip/hip_runtime.h>

// ---------------------------------------------------------------------------
// LISTA recurrence for MI455X (gfx1250), wave32 + WMMA bf16.
//
// Dims (fixed by reference):
//   T=50, B=256, N_IN=1024, N_HID=4096, N_COMP=256, K_ITERS=3
//
// Math rewrite (rank-256 structure):
//   W = G - U@ADG  =>  h0@W = h0@G - (h0@U)@ADG     (precompute "base0" once)
//   S = I - U@AD   =>  h@S  = h  - (h@U)@AD         (2x 256-K GEMMs, not 4096)
//
// Wave->tile mapping: waves within a workgroup share the SAME B strip
// (wm = wave % wavesM). A workgroup runs on one WGP, so the 8 identical
// B-fragment load streams coalesce in WGP$ and each fragment of the big
// streamed operand (G, D^T) is pulled from L2 exactly once per GEMM.
// ---------------------------------------------------------------------------

#define T_STEPS 50
#define BATCH   256
#define NIN     1024
#define NHID    4096
#define NCOMP   256

typedef __attribute__((ext_vector_type(16))) __bf16 bf16x16;
typedef __attribute__((ext_vector_type(8)))  float  f32x8;

// ---------------------------------------------------------------------------
// Packing index math for the CDNA5 WMMA 16x16x32 bf16 fragment layouts.
// A-fragment (16x32 tile): lane = h*16 + (m%16);  K-in-tile = (e<8?0:16)+8h+(e&7)
//   inverse: h = (kk>>3)&1 ; e = (kk&7) + ((kk&16)>>1)
// B-fragment (32x16 tile): lane = h*16 + (n%16);  K-in-tile = (e<8?0:16)+8h+(e&7)
// Tiles stored [tileIdx][lane][e] contiguously => per-lane 32B contiguous loads.
// ---------------------------------------------------------------------------

__device__ __forceinline__ size_t afrag_off(int m, int k, int K) {
  int mt = m >> 4, kt = k >> 5, kk = k & 31;
  int h  = (kk >> 3) & 1;
  int e  = (kk & 7) + ((kk & 16) >> 1);
  int lane = (h << 4) + (m & 15);
  return ((size_t)(mt * (K >> 5) + kt) * 32 + lane) * 16 + e;
}

__device__ __forceinline__ size_t bfrag_off(int k, int n, int K) {
  int kt = k >> 5, nt = n >> 4, kk = k & 31;
  int h  = (kk >> 3) & 1;
  int e  = (kk & 7) + ((kk & 16) >> 1);
  int lane = (h << 4) + (n & 15);
  return ((size_t)(nt * (K >> 5) + kt) * 32 + lane) * 16 + e;
}

// ---------------------------------------------------------------------------
// Pack row-major f32 (optionally transposed) -> bf16 A-fragment layout
// value(m,k) = trans ? src[k*ld + m] : src[m*ld + k]
// ---------------------------------------------------------------------------
__global__ void lista_pack_a(const float* __restrict__ src, __bf16* __restrict__ dst,
                             int M, int K, int ld, int trans) {
  int idx = blockIdx.x * blockDim.x + threadIdx.x;
  if (idx >= M * K) return;
  int m = idx / K, k = idx - m * K;
  float v = trans ? src[(size_t)k * ld + m] : src[(size_t)m * ld + k];
  dst[afrag_off(m, k, K)] = (__bf16)v;
}

// value(k,n) = trans ? src[n*ld + k] : src[k*ld + n]
__global__ void lista_pack_b(const float* __restrict__ src, __bf16* __restrict__ dst,
                             int K, int N, int ld, int trans) {
  int idx = blockIdx.x * blockDim.x + threadIdx.x;
  if (idx >= K * N) return;
  int k = idx / N, n = idx - k * N;
  float v = trans ? src[(size_t)n * ld + k] : src[(size_t)k * ld + n];
  dst[bfrag_off(k, n, K)] = (__bf16)v;
}

// ---------------------------------------------------------------------------
// WMMA GEMM: Cout = alpha*(A@B) + add1 + add2  (adds optional)
//   alpha = alpha_c * (alpha_recip ? 1/alpha_recip[0] : 1)
// A: packed A-frag bf16, M x K.  B: packed B-frag bf16, K x N.  C: f32 row-major.
// Per-wave macro tile 32x64 (2x4 WMMA tiles). Requires M%32==0, N%64==0, K%32==0.
// ---------------------------------------------------------------------------
__device__ __forceinline__ void lista_store_tile(float* __restrict__ Cout, const f32x8& t,
                                                 int mBase, int n, int N,
                                                 const float* __restrict__ add1,
                                                 const float* __restrict__ add2,
                                                 float alpha) {
#pragma unroll
  for (int r = 0; r < 8; ++r) {
    size_t idx = (size_t)(mBase + r) * N + n;
    float x = alpha * t[r];
    if (add1) x += add1[idx];
    if (add2) x += add2[idx];
    Cout[idx] = x;
  }
}

__global__ __launch_bounds__(256)
void lista_gemm_bf16(const __bf16* __restrict__ Apk,
                     const __bf16* __restrict__ Bpk,
                     float* __restrict__ Cout,
                     const float* __restrict__ add1,
                     const float* __restrict__ add2,
                     const float* __restrict__ alpha_recip,
                     float alpha_c,
                     int M, int N, int K) {
  const int lane = threadIdx.x & 31;
  const int wave = blockIdx.x * (blockDim.x >> 5) + (threadIdx.x >> 5);
  const int wavesM = M >> 5;
  const int wavesN = N >> 6;
  const int totalWaves = wavesM * wavesN;
  if (wave >= totalWaves) return;
  // waves of one block advance along M and share the B strip (WGP$ reuse);
  // the big streamed B operand is then fetched from L2 once per GEMM.
  const int wm = wave % wavesM;
  const int wn = wave / wavesM;
  const int KT = K >> 5;

  f32x8 acc00 = (f32x8)0.0f, acc01 = (f32x8)0.0f, acc02 = (f32x8)0.0f, acc03 = (f32x8)0.0f;
  f32x8 acc10 = (f32x8)0.0f, acc11 = (f32x8)0.0f, acc12 = (f32x8)0.0f, acc13 = (f32x8)0.0f;

  const __bf16* pa0 = Apk + ((size_t)(2 * wm + 0) * KT * 32 + lane) * 16;
  const __bf16* pa1 = Apk + ((size_t)(2 * wm + 1) * KT * 32 + lane) * 16;
  const __bf16* pb0 = Bpk + ((size_t)(4 * wn + 0) * KT * 32 + lane) * 16;
  const __bf16* pb1 = Bpk + ((size_t)(4 * wn + 1) * KT * 32 + lane) * 16;
  const __bf16* pb2 = Bpk + ((size_t)(4 * wn + 2) * KT * 32 + lane) * 16;
  const __bf16* pb3 = Bpk + ((size_t)(4 * wn + 3) * KT * 32 + lane) * 16;

  for (int kt = 0; kt < KT; ++kt) {
    bf16x16 a0 = *(const bf16x16*)pa0;
    bf16x16 a1 = *(const bf16x16*)pa1;
    bf16x16 b0 = *(const bf16x16*)pb0;
    bf16x16 b1 = *(const bf16x16*)pb1;
    bf16x16 b2 = *(const bf16x16*)pb2;
    bf16x16 b3 = *(const bf16x16*)pb3;
    // prefetch 4 k-tiles ahead on the streaming operands (speculative, safe)
    __builtin_prefetch(pb0 + 2048, 0, 1);
    __builtin_prefetch(pb2 + 2048, 0, 1);
    __builtin_prefetch(pa0 + 2048, 0, 1);
    pa0 += 512; pa1 += 512; pb0 += 512; pb1 += 512; pb2 += 512; pb3 += 512;

    acc00 = __builtin_amdgcn_wmma_f32_16x16x32_bf16(false, a0, false, b0, (short)0, acc00, false, false);
    acc01 = __builtin_amdgcn_wmma_f32_16x16x32_bf16(false, a0, false, b1, (short)0, acc01, false, false);
    acc02 = __builtin_amdgcn_wmma_f32_16x16x32_bf16(false, a0, false, b2, (short)0, acc02, false, false);
    acc03 = __builtin_amdgcn_wmma_f32_16x16x32_bf16(false, a0, false, b3, (short)0, acc03, false, false);
    acc10 = __builtin_amdgcn_wmma_f32_16x16x32_bf16(false, a1, false, b0, (short)0, acc10, false, false);
    acc11 = __builtin_amdgcn_wmma_f32_16x16x32_bf16(false, a1, false, b1, (short)0, acc11, false, false);
    acc12 = __builtin_amdgcn_wmma_f32_16x16x32_bf16(false, a1, false, b2, (short)0, acc12, false, false);
    acc13 = __builtin_amdgcn_wmma_f32_16x16x32_bf16(false, a1, false, b3, (short)0, acc13, false, false);
  }

  float alpha = alpha_c;
  if (alpha_recip) alpha *= 1.0f / alpha_recip[0];

  // C tile layout: VGPR r, lanes 0-15 -> M=r, lanes 16-31 -> M=8+r; N = lane%16
  const int nb  = (wn << 6) + (lane & 15);
  const int mb0 = (wm << 5) + ((lane >> 4) << 3);
  const int mb1 = mb0 + 16;
  lista_store_tile(Cout, acc00, mb0, nb + 0,  N, add1, add2, alpha);
  lista_store_tile(Cout, acc01, mb0, nb + 16, N, add1, add2, alpha);
  lista_store_tile(Cout, acc02, mb0, nb + 32, N, add1, add2, alpha);
  lista_store_tile(Cout, acc03, mb0, nb + 48, N, add1, add2, alpha);
  lista_store_tile(Cout, acc10, mb1, nb + 0,  N, add1, add2, alpha);
  lista_store_tile(Cout, acc11, mb1, nb + 16, N, add1, add2, alpha);
  lista_store_tile(Cout, acc12, mb1, nb + 32, N, add1, add2, alpha);
  lista_store_tile(Cout, acc13, mb1, nb + 48, N, add1, add2, alpha);
}

// ---------------------------------------------------------------------------
// phi: piecewise soft-threshold (sequential overriding masks, as in reference)
//   u = u1 + (u2 ? u2 : 0)
// Writes h both as row-major f32 and as packed bf16 A-fragment (M x K).
// ---------------------------------------------------------------------------
__global__ void lista_phi(const float* __restrict__ u1, const float* __restrict__ u2,
                          const float* __restrict__ vsrc,
                          const float* __restrict__ l1p, const float* __restrict__ l2p,
                          const float* __restrict__ ap,
                          float* __restrict__ h_f32, __bf16* __restrict__ h_pk,
                          int M, int K) {
  int idx = blockIdx.x * blockDim.x + threadIdx.x;
  if (idx >= M * K) return;
  float a  = ap[0];
  float g1 = l1p[0] / a;
  float g2 = l2p[0] / a;
  float u = u1[idx] + (u2 ? u2[idx] : 0.0f);
  float v = vsrc[idx];
  bool vp = (v >= 0.0f), vn = !vp;
  float out = 0.0f;
  if ((vp && (v + g1 + g2 <= u)) || (vn && (u >= g1 + g2)))                                   out = u - g1 - g2;
  if ((vp && (v + g1 - g2 <= u) && (u < v + g1 + g2)) ||
      (vn && (v - g1 + g2 > u)  && (u >= v - g1 - g2)))                                       out = v;
  if ((vp && (g1 - g2 <= u) && (v + g1 - g2 > u)) || (vn && (u < v - g1 - g2)))               out = u - g1 + g2;
  if ((vp && (u >= -g1 - g2) && (u < g1 - g2)) || (vn && (u >= -g1 + g2) && (u < g1 + g2)))   out = 0.0f;
  if (vp && (u < -g1 - g2))                                                                   out = u + g1 + g2;
  if (vn && (u < v - g1 - g2))                                                                out = u - g1 + g2;
  h_f32[idx] = out;
  int m = idx / K, k = idx - m * K;
  h_pk[afrag_off(m, k, K)] = (__bf16)out;
}

// ---------------------------------------------------------------------------
// Host driver
// ---------------------------------------------------------------------------
extern "C" void kernel_launch(void* const* d_in, const int* in_sizes, int n_in,
                              void* d_out, int out_size, void* d_ws, size_t ws_size,
                              hipStream_t stream) {
  (void)in_sizes; (void)n_in; (void)out_size; (void)ws_size;

  const float* data = (const float*)d_in[0];   // (T,B,NIN)
  const float* A    = (const float*)d_in[1];   // (NCOMP,NIN)
  const float* Dm   = (const float*)d_in[2];   // (NIN,NHID)
  const float* h0   = (const float*)d_in[3];   // (B,NHID)
  const float* G    = (const float*)d_in[4];   // (NHID,NHID)
  const float* l1   = (const float*)d_in[5];
  const float* l2   = (const float*)d_in[6];
  const float* aSc  = (const float*)d_in[7];
  float* out = (float*)d_out;                  // (T,B,NIN)

  char* wsb = (char*)d_ws;
  size_t off = 0;
  auto alloc = [&](size_t bytes) -> void* {
    void* p = wsb + off;
    off += (bytes + 255) & ~(size_t)255;
    return p;
  };

  // persistent packed operands (bf16)
  __bf16* AT_B  = (__bf16*)alloc((size_t)NIN  * NCOMP * 2);  // A^T as B-op (K=NIN,N=NCOMP)
  __bf16* A_A   = (__bf16*)alloc((size_t)NCOMP* NIN   * 2);  // A as A-op
  __bf16* D_B   = (__bf16*)alloc((size_t)NIN  * NHID  * 2);  // D as B-op
  __bf16* DT_A  = (__bf16*)alloc((size_t)NHID * NIN   * 2);  // D^T as A-op
  __bf16* DT_B  = (__bf16*)alloc((size_t)NHID * NIN   * 2);  // D^T as B-op (K=NHID,N=NIN)
  __bf16* G_B   = (__bf16*)alloc((size_t)NHID * NHID  * 2);  // G as B-op
  float*  U_f   = (float*) alloc((size_t)NHID * NCOMP * 4);  // U = (1/a) D^T A^T
  __bf16* U_B   = (__bf16*)alloc((size_t)NHID * NCOMP * 2);  // U as B-op   (K=NHID,N=NCOMP)
  __bf16* UT_B  = (__bf16*)alloc((size_t)NCOMP* NHID  * 2);  // U^T as B-op (K=NCOMP,N=NHID)
  float*  AD_f  = (float*) alloc((size_t)NCOMP* NHID  * 4);  // AD = A@D
  __bf16* AD_B  = (__bf16*)alloc((size_t)NCOMP* NHID  * 2);  // AD as B-op  (K=NCOMP,N=NHID)
  __bf16* AD_A  = (__bf16*)alloc((size_t)NCOMP* NHID  * 2);  // AD as A-op  (M=NCOMP,K=NHID)
  float*  ADG_f = (float*) alloc((size_t)NCOMP* NHID  * 4);  // ADG = AD@G
  __bf16* ADG_B = (__bf16*)alloc((size_t)NCOMP* NHID  * 2);  // ADG as B-op (K=NCOMP,N=NHID)
  __bf16* h0_A  = (__bf16*)alloc((size_t)BATCH* NHID  * 2);
  float*  hg0   = (float*) alloc((size_t)BATCH* NHID  * 4);  // h0@G (also v for t=0)
  float*  base0 = (float*) alloc((size_t)BATCH* NHID  * 4);  // h0@W
  // per-step buffers
  __bf16* dt_A  = (__bf16*)alloc((size_t)BATCH* NIN   * 2);
  float*  cT_f  = (float*) alloc((size_t)BATCH* NCOMP * 4);
  __bf16* cT_A  = (__bf16*)alloc((size_t)BATCH* NCOMP * 2);
  float*  cu    = (float*) alloc((size_t)BATCH* NHID  * 4);
  float*  v_buf = (float*) alloc((size_t)BATCH* NHID  * 4);
  float*  q_buf = (float*) alloc((size_t)BATCH* NHID  * 4);
  float*  h_f   = (float*) alloc((size_t)BATCH* NHID  * 4);
  __bf16* h_A   = (__bf16*)alloc((size_t)BATCH* NHID  * 2);
  float*  p_f   = (float*) alloc((size_t)BATCH* NCOMP * 4);
  __bf16* p_A   = (__bf16*)alloc((size_t)BATCH* NCOMP * 2);

  auto packA = [&](const float* src, __bf16* dst, int M, int K, int ld, int trans) {
    int n = M * K;
    lista_pack_a<<<(n + 255) / 256, 256, 0, stream>>>(src, dst, M, K, ld, trans);
  };
  auto packB = [&](const float* src, __bf16* dst, int K, int N, int ld, int trans) {
    int n = K * N;
    lista_pack_b<<<(n + 255) / 256, 256, 0, stream>>>(src, dst, K, N, ld, trans);
  };
  auto gemm = [&](const __bf16* Ap, const __bf16* Bp, float* Co,
                  const float* add1, const float* add2,
                  const float* recip, float ac, int M, int N, int K) {
    int waves  = (M >> 5) * (N >> 6);
    int blocks = (waves + 7) >> 3;
    lista_gemm_bf16<<<blocks, 256, 0, stream>>>(Ap, Bp, Co, add1, add2, recip, ac, M, N, K);
  };
  auto phiL = [&](const float* u1, const float* u2, const float* v,
                  float* hf, __bf16* hpk, int M, int K) {
    int n = M * K;
    lista_phi<<<(n + 255) / 256, 256, 0, stream>>>(u1, u2, v, l1, l2, aSc, hf, hpk, M, K);
  };

  // ---------------- precompute ----------------
  packB(A,  AT_B, NIN,  NCOMP, NIN,  1);   // A^T[k][n] = A[n][k]
  packA(A,  A_A,  NCOMP, NIN,  NIN,  0);
  packB(Dm, D_B,  NIN,  NHID,  NHID, 0);
  packA(Dm, DT_A, NHID, NIN,   NHID, 1);   // D^T[m][k] = D[k][m]
  packB(Dm, DT_B, NHID, NIN,   NHID, 1);   // D^T[k][n] = D[n][k]
  packB(G,  G_B,  NHID, NHID,  NHID, 0);

  gemm(DT_A, AT_B, U_f, nullptr, nullptr, aSc, 1.0f, NHID, NCOMP, NIN);  // U=(1/a)D^T A^T
  packB(U_f, U_B,  NHID,  NCOMP, NCOMP, 0);
  packB(U_f, UT_B, NCOMP, NHID,  NCOMP, 1);

  gemm(A_A, D_B, AD_f, nullptr, nullptr, nullptr, 1.0f, NCOMP, NHID, NIN); // AD=A@D
  packB(AD_f, AD_B, NCOMP, NHID, NHID, 0);
  packA(AD_f, AD_A, NCOMP, NHID, NHID, 0);

  gemm(AD_A, G_B, ADG_f, nullptr, nullptr, nullptr, 1.0f, NCOMP, NHID, NHID); // ADG=AD@G
  packB(ADG_f, ADG_B, NCOMP, NHID, NHID, 0);

  packA(h0, h0_A, BATCH, NHID, NHID, 0);
  gemm(h0_A, G_B, hg0, nullptr, nullptr, nullptr, 1.0f, BATCH, NHID, NHID);  // h0@G
  gemm(h0_A, U_B, p_f, nullptr, nullptr, nullptr, 1.0f, BATCH, NCOMP, NHID); // h0@U
  packA(p_f, p_A, BATCH, NCOMP, NCOMP, 0);
  // base0 = h0@W = hg0 - (h0@U)@ADG
  gemm(p_A, ADG_B, base0, hg0, nullptr, nullptr, -1.0f, BATCH, NHID, NCOMP);

  // ---------------- time loop ----------------
  const float* vptr = hg0;   // v for t=0 is h0@G
  for (int t = 0; t < T_STEPS; ++t) {
    // c_t = data_t @ A^T ; cu = c_t @ U^T
    packA(data + (size_t)t * BATCH * NIN, dt_A, BATCH, NIN, NIN, 0);
    gemm(dt_A, AT_B, cT_f, nullptr, nullptr, nullptr, 1.0f, BATCH, NCOMP, NIN);
    packA(cT_f, cT_A, BATCH, NCOMP, NCOMP, 0);
    gemm(cT_A, UT_B, cu, nullptr, nullptr, nullptr, 1.0f, BATCH, NHID, NCOMP);

    // h1 = phi(base0 + cu, v)
    phiL(base0, cu, vptr, h_f, h_A, BATCH, NHID);

    // two LISTA iterations: h = phi(h - (h@U)@AD + cu, v)
    for (int it = 0; it < 2; ++it) {
      gemm(h_A, U_B, p_f, nullptr, nullptr, nullptr, 1.0f, BATCH, NCOMP, NHID);
      packA(p_f, p_A, BATCH, NCOMP, NCOMP, 0);
      gemm(p_A, AD_B, q_buf, h_f, cu, nullptr, -1.0f, BATCH, NHID, NCOMP);
      phiL(q_buf, nullptr, vptr, h_f, h_A, BATCH, NHID);
    }

    // s_t = h3 @ D^T  -> output
    gemm(h_A, DT_B, out + (size_t)t * BATCH * NIN, nullptr, nullptr, nullptr, 1.0f,
         BATCH, NIN, NHID);

    // v_{t+1} = h3 @ G
    if (t + 1 < T_STEPS) {
      gemm(h_A, G_B, v_buf, nullptr, nullptr, nullptr, 1.0f, BATCH, NHID, NHID);
      vptr = v_buf;
    }
  }
}
